// BorderBranch_66606352827277
// MI455X (gfx1250) — matrix operand order repaired and checked by reading, not compiled, not run on hardware
//
#include <hip/hip_runtime.h>
#include <hip/hip_bf16.h>

typedef __bf16 bf16;
typedef __bf16 v16bf __attribute__((ext_vector_type(16)));
typedef float  v8f   __attribute__((ext_vector_type(8)));
typedef unsigned int u32x4 __attribute__((ext_vector_type(4)));
typedef int i32x4 __attribute__((ext_vector_type(4)));
typedef int i32x8 __attribute__((ext_vector_type(8)));

union Frag { u32x4 q[2]; v16bf v; };

constexpr int NB  = 2;
constexpr int CIN = 256;
constexpr int H = 96, W = 96, HW = H * W;   // 9216
constexpr int BC = 128;
constexpr int C5 = 5 * BC;                  // 640
constexpr int C4 = 4 * BC;                  // 512
constexpr int MT  = HW / 16;                // 576 M-tiles per batch
constexpr int KS1 = CIN / 32;               // 8
constexpr int KS2 = 9 * C4 / 32;            // 144
constexpr int KS3 = C5 / 32;                // 20
constexpr int NT1 = C5 / 16;                // 40
constexpr int NT3 = 256 / 16;               // 16

// ---- workspace layout (bytes, all chunks 256B-aligned) ----
constexpr size_t OFF_FEATA   = 0;                                      // [NB][HW][CIN] bf16
constexpr size_t OFF_B1      = OFF_FEATA   + (size_t)NB * HW * CIN * 2;
constexpr size_t OFF_BMASK   = OFF_B1      + (size_t)NT1 * KS1 * 1024; // 1024B per (nt,ks)
constexpr size_t OFF_BBORDER = OFF_BMASK   + (size_t)NT1 * KS2 * 1024;
constexpr size_t OFF_C1      = OFF_BBORDER + (size_t)NT3 * KS3 * 1024; // [NB][HW][C5] f32 (aliased by Afinal)
constexpr size_t OFF_STATS   = OFF_C1      + (size_t)NB * HW * C5 * 4;
constexpr size_t OFF_FEAT4   = OFF_STATS   + (size_t)NB * C5 * 2 * 4;  // [NB][HW][C4] bf16
constexpr size_t OFF_ALIGN   = OFF_FEAT4   + (size_t)NB * HW * C4 * 2; // [NB][HW][C5] bf16

// ---------------- NCHW f32 -> [N][HW][C] bf16 (LDS tiled transpose) ----------------
__global__ void k_feat_to_bf16(const float* __restrict__ feat, bf16* __restrict__ A) {
  __shared__ float tile[32][33];
  int n = blockIdx.z;
  int hw0 = blockIdx.x * 32;
  int c0  = blockIdx.y * 32;
  int tx = threadIdx.x, ty = threadIdx.y;          // 32 x 8
#pragma unroll
  for (int i = 0; i < 4; ++i) {
    int c = c0 + ty + i * 8;
    tile[ty + i * 8][tx] = feat[((size_t)n * CIN + c) * HW + hw0 + tx];
  }
  __syncthreads();
#pragma unroll
  for (int i = 0; i < 4; ++i) {
    int hw = hw0 + ty + i * 8;
    A[((size_t)n * HW + hw) * CIN + c0 + tx] = (bf16)tile[tx][ty + i * 8];
  }
}

// ---------------- weight fragment packing ----------------
// B packing used here (contiguous k order, same as the A loads): lane l -> col n = l%16; K = (l/16)*16 + 2v + e.
__global__ void k_pack_b1(const float* __restrict__ w_cur, const float* __restrict__ w_ltrb,
                          bf16* __restrict__ Bp) {
  int tid = blockIdx.x * blockDim.x + threadIdx.x;
  if (tid >= NT1 * KS1 * 32 * 8) return;
  int v = tid & 7;
  int lane = (tid >> 3) & 31;
  int ks = (tid >> 8) % KS1;
  int nt = tid / (256 * KS1);
  int col = nt * 16 + (lane & 15);
  int k0 = ks * 32 + (lane >> 4) * 16 + 2 * v;
  float w0, w1;
  if (col < BC) { w0 = w_cur[(size_t)col * CIN + k0];         w1 = w_cur[(size_t)col * CIN + k0 + 1]; }
  else          { w0 = w_ltrb[(size_t)(col - BC) * CIN + k0]; w1 = w_ltrb[(size_t)(col - BC) * CIN + k0 + 1]; }
  Bp[(size_t)tid * 2]     = (bf16)w0;
  Bp[(size_t)tid * 2 + 1] = (bf16)w1;
}

__global__ void k_pack_bmask(const float* __restrict__ w_mask, bf16* __restrict__ Bp) {
  int tid = blockIdx.x * blockDim.x + threadIdx.x;
  if (tid >= NT1 * KS2 * 32 * 8) return;
  int v = tid & 7;
  int lane = (tid >> 3) & 31;
  int ks = (tid >> 8) % KS2;
  int nt = tid / (256 * KS2);
  int oc = nt * 16 + (lane & 15);
  int t = ks / 16, kk = ks % 16;                 // tap, cin-chunk
  int cin0 = kk * 32 + (lane >> 4) * 16 + 2 * v;
  int ky = t / 3, kx = t % 3;
  size_t base = (((size_t)oc * C4 + cin0) * 3 + ky) * 3 + kx;   // w_mask[oc][cin][ky][kx]
  Bp[(size_t)tid * 2]     = (bf16)w_mask[base];
  Bp[(size_t)tid * 2 + 1] = (bf16)w_mask[base + 9];             // cin+1 -> +3*3
}

__global__ void k_pack_bborder(const float* __restrict__ w_border, bf16* __restrict__ Bp) {
  int tid = blockIdx.x * blockDim.x + threadIdx.x;
  if (tid >= NT3 * KS3 * 32 * 8) return;
  int v = tid & 7;
  int lane = (tid >> 3) & 31;
  int ks = (tid >> 8) % KS3;
  int nt = tid / (256 * KS3);
  int oc = nt * 16 + (lane & 15);
  int k0 = ks * 32 + (lane >> 4) * 16 + 2 * v;
  Bp[(size_t)tid * 2]     = (bf16)w_border[(size_t)oc * C5 + k0];
  Bp[(size_t)tid * 2 + 1] = (bf16)w_border[(size_t)oc * C5 + k0 + 1];
}

// ---------------- GEMM1: C1[n][hw][640] = featA[n][hw][256] x B1 ----------------
__global__ void __launch_bounds__(32) k_gemm1(const bf16* __restrict__ A,
                                              const bf16* __restrict__ Bp,
                                              float* __restrict__ C) {
  int lane = threadIdx.x & 31;
  int ng = blockIdx.y;                  // 0..9 -> nt = ng*4 + j
  int mt = blockIdx.x % MT;
  int n  = blockIdx.x / MT;
  int half = lane >> 4, mrow = lane & 15;
  const u32x4* arow = (const u32x4*)(A + ((size_t)n * HW + mt * 16 + mrow) * CIN);
  v8f acc[4] = {v8f{}, v8f{}, v8f{}, v8f{}};
  Frag a[2], b[2][4];
  auto load_step = [&](int ks, int buf) {
    const u32x4* ap = arow + ks * 4;
    a[buf].q[0] = ap[2 * half];
    a[buf].q[1] = ap[2 * half + 1];
#pragma unroll
    for (int j = 0; j < 4; ++j) {
      const u32x4* bp2 = (const u32x4*)Bp + ((size_t)((ng * 4 + j) * KS1 + ks) * 32 + lane) * 2;
      b[buf][j].q[0] = bp2[0];
      b[buf][j].q[1] = bp2[1];
    }
  };
  load_step(0, 0);
#pragma unroll
  for (int ks = 0; ks < KS1; ++ks) {
    int cur = ks & 1;
    if (ks + 1 < KS1) load_step(ks + 1, cur ^ 1);
#pragma unroll
    for (int j = 0; j < 4; ++j)
      acc[j] = __builtin_amdgcn_wmma_f32_16x16x32_bf16(false, a[cur].v, false, b[cur][j].v,
                                                       (short)0, acc[j], false, false);
  }
#pragma unroll
  for (int j = 0; j < 4; ++j) {
    float* crow = C + ((size_t)n * HW + mt * 16) * C5 + (ng * 4 + j) * 16 + mrow;
#pragma unroll
    for (int v = 0; v < 8; ++v)
      crow[(size_t)(v + 8 * half) * C5] = acc[j][v];
  }
}

// ---------------- instance-norm stats per (n, ch) ----------------
__global__ void k_stats(const float* __restrict__ C, float* __restrict__ stats) {
  int ch = blockIdx.x % C5;
  int n  = blockIdx.x / C5;
  int t = threadIdx.x;
  float s = 0.f, s2 = 0.f;
  for (int i = t; i < HW; i += 256) {
    float x = C[((size_t)n * HW + i) * C5 + ch];
    s += x; s2 += x * x;
  }
  __shared__ float sh[256], sh2[256];
  sh[t] = s; sh2[t] = s2;
  __syncthreads();
  for (int o = 128; o > 0; o >>= 1) {
    if (t < o) { sh[t] += sh[t + o]; sh2[t] += sh2[t + o]; }
    __syncthreads();
  }
  if (t == 0) {
    float m = sh[0] / (float)HW;
    float var = sh2[0] / (float)HW - m * m;
    stats[(size_t)blockIdx.x * 2]     = m;
    stats[(size_t)blockIdx.x * 2 + 1] = rsqrtf(var + 1e-5f);
  }
}

// ---------------- normalize + relu -> feat4 bf16 and fm_short slice of align ----------------
__global__ void k_norm(const float* __restrict__ C, const float* __restrict__ stats,
                       bf16* __restrict__ align, bf16* __restrict__ feat4) {
  size_t idx = (size_t)blockIdx.x * blockDim.x + threadIdx.x;
  if (idx >= (size_t)NB * HW * C5) return;
  int ch = (int)(idx % C5);
  size_t nhw = idx / C5;
  int n = (int)(nhw / HW);
  float m = stats[((size_t)n * C5 + ch) * 2];
  float r = stats[((size_t)n * C5 + ch) * 2 + 1];
  float v = fmaxf((C[idx] - m) * r, 0.f);
  if (ch < BC) align[nhw * C5 + C4 + ch] = (bf16)v;       // fm_short -> align channels 512..639
  else         feat4[nhw * C4 + (ch - BC)] = (bf16)v;     // feat4 channels 0..511
}

// ---------------- border align: max over 11 bilinear samples per border ----------------
__global__ void k_border(const bf16* __restrict__ feat4, const float* __restrict__ boxes,
                         bf16* __restrict__ align) {
  int x = blockIdx.x;
  int b  = x & 3;
  int hw = (x >> 2) % HW;
  int n  = (x >> 2) / HW;
  int c = threadIdx.x;                                     // 0..127
  const float* bx = boxes + ((size_t)n * HW + hw) * 4;
  float x1 = bx[0], y1 = bx[1], x2 = bx[2], y2 = bx[3];
  float bw = x2 - x1, bh = y2 - y1;
  float sx, sy, dx, dy;
  if      (b == 0) { sx = x1; sy = y1; dx = bw * 0.1f; dy = 0.f; }        // top
  else if (b == 1) { sx = x1; sy = y1; dx = 0.f;       dy = bh * 0.1f; }  // left
  else if (b == 2) { sx = x1; sy = y2; dx = bw * 0.1f; dy = 0.f; }        // bottom
  else             { sx = x2; sy = y1; dx = 0.f;       dy = bh * 0.1f; }  // right
  const bf16* f = feat4 + (size_t)n * HW * C4 + b * BC + c;
  float best = -INFINITY;
  for (int i = 0; i <= 10; ++i) {
    float px = sx + dx * (float)i;
    float py = sy + dy * (float)i;
    bool valid = (px >= -1.f) && (px <= (float)W) && (py >= -1.f) && (py <= (float)H);
    float xc = fminf(fmaxf(px, 0.f), (float)(W - 1));
    float yc = fminf(fmaxf(py, 0.f), (float)(H - 1));
    int x0 = min((int)floorf(xc), W - 1);
    int y0 = min((int)floorf(yc), H - 1);
    int x1i = min(x0 + 1, W - 1);
    int y1i = min(y0 + 1, H - 1);
    float lx = (x0 >= W - 1) ? 0.f : xc - (float)x0;
    float ly = (y0 >= H - 1) ? 0.f : yc - (float)y0;
    float hx = 1.f - lx, hy = 1.f - ly;
    float v00 = (float)f[((size_t)y0  * W + x0 ) * C4];
    float v01 = (float)f[((size_t)y0  * W + x1i) * C4];
    float v10 = (float)f[((size_t)y1i * W + x0 ) * C4];
    float v11 = (float)f[((size_t)y1i * W + x1i) * C4];
    float val = v00 * hy * hx + v01 * hy * lx + v10 * ly * hx + v11 * ly * lx;
    val = valid ? val : 0.f;
    best = fmaxf(best, val);
  }
  // reference's transpose(0,3,1,2).reshape(N,4BC,H,W) channel scrambling
  int fi = hw * 4 + b;
  int och  = c * 4 + fi / HW;
  int opix = fi % HW;
  align[((size_t)n * HW + opix) * C5 + och] = (bf16)best;
}

// ---------------- conv3x3: TDM-staged B in LDS + implicit-im2col WMMA ----------------
// 4 waves/workgroup, each wave a 2Mx4N tile (same 64 out-channels). Per tap, wave 0
// issues one tensor_load_to_lds DMA of the 4x16KB packed-B block (2D tile, row stride
// KS2*1024B); waves read B fragments from LDS (ds_load_b128), A from global.
__global__ void __launch_bounds__(128) k_conv3x3(const bf16* __restrict__ feat4,
                                                 const bf16* __restrict__ Bp,
                                                 const float* __restrict__ bmask,
                                                 const bf16* __restrict__ align,
                                                 bf16* __restrict__ afinal) {
  __shared__ __align__(16) unsigned char smemB[65536];   // 4 nt x 16 kk x 1024B
  int wave = threadIdx.x >> 5;
  int lane = threadIdx.x & 31;
  int ng = blockIdx.y;                       // 0..9 -> nt = ng*4 + j
  int mg = blockIdx.x % ((MT / 2) / 4);      // 0..71
  int n  = blockIdx.x / ((MT / 2) / 4);
  int mp = mg * 4 + wave;                    // 0..287: y*3 + xpair
  int y = mp / 3, x0 = (mp % 3) * 32;        // 32-pixel double M tile within one row
  int half = lane >> 4, mrow = lane & 15;
  v8f acc[8] = {v8f{}, v8f{}, v8f{}, v8f{}, v8f{}, v8f{}, v8f{}, v8f{}};
  const u32x4 zq = {0u, 0u, 0u, 0u};
  unsigned ldsBase = (unsigned)(uintptr_t)&smemB[0];
#pragma unroll 1
  for (int t = 0; t < 9; ++t) {
    __syncthreads();                         // previous tap fully consumed
    if (wave == 0) {
      // ---- TDM descriptor: 2D tile, 4096 dwords x 4 rows, row stride 36864 dwords ----
      unsigned long long ga = (unsigned long long)(uintptr_t)
          ((const unsigned char*)Bp + ((size_t)(ng * 4) * KS2 + (size_t)t * 16) * 1024);
      u32x4 g0;
      g0.x = 1u;                                             // count=1, user descriptor
      g0.y = ldsBase;                                        // lds_addr
      g0.z = (unsigned)(ga & 0xFFFFFFFFu);                   // global_addr[31:0]
      g0.w = (unsigned)((ga >> 32) & 0x1FFFFFFu) | (2u << 30); // global_addr[56:32] | type=2
      i32x8 g1;
      g1[0] = (int)(2u << 16);                               // data_size=4B
      g1[1] = (int)(4096u << 16);                            // tensor_dim0[15:0]=4096
      g1[2] = (int)(4u << 16);                               // tensor_dim0 hi=0 | tensor_dim1=4
      g1[3] = (int)(4096u << 16);                            // tensor_dim1 hi=0 | tile_dim0=4096
      g1[4] = 4;                                             // tile_dim1=4, tile_dim2=0
      g1[5] = 36864;                                         // tensor_dim0_stride[31:0] (dwords)
      g1[6] = 0;                                             // stride0 hi | stride1 lo
      g1[7] = 0;
      i32x4 gz4 = {0, 0, 0, 0};
      i32x8 gz8 = {0, 0, 0, 0, 0, 0, 0, 0};
      __builtin_amdgcn_tensor_load_to_lds(g0, g1, gz4, gz4, gz8, 0);
      __builtin_amdgcn_s_wait_tensorcnt(0);
    }
    __syncthreads();                         // B tile visible to all waves

    int dy = t / 3 - 1, dx = t % 3 - 1;
    int py = y + dy;
    bool rowok = (py >= 0) && (py < H);
    int px0 = x0 + mrow + dx;
    int px1 = px0 + 16;
    bool ok0 = rowok && (px0 >= 0) && (px0 < W);
    bool ok1 = rowok && (px1 < W);           // px1 >= 15 always
    const u32x4* ap0 = (const u32x4*)(feat4 + ((size_t)n * HW + py * W + px0) * C4);
    const u32x4* ap1 = (const u32x4*)(feat4 + ((size_t)n * HW + py * W + px1) * C4);
    Frag a0[2], a1[2];
    auto load_a = [&](int kk, int buf) {
      if (ok0) { const u32x4* p = ap0 + kk * 4; a0[buf].q[0] = p[2 * half]; a0[buf].q[1] = p[2 * half + 1]; }
      else     { a0[buf].q[0] = zq; a0[buf].q[1] = zq; }
      if (ok1) { const u32x4* p = ap1 + kk * 4; a1[buf].q[0] = p[2 * half]; a1[buf].q[1] = p[2 * half + 1]; }
      else     { a1[buf].q[0] = zq; a1[buf].q[1] = zq; }
    };
    load_a(0, 0);
#pragma unroll
    for (int kk = 0; kk < 16; ++kk) {
      int cur = kk & 1;
      if (kk + 1 < 16) load_a(kk + 1, cur ^ 1);
#pragma unroll
      for (int j = 0; j < 4; ++j) {
        Frag b;
        const u32x4* bls = (const u32x4*)(smemB + (size_t)(j * 16 + kk) * 1024) + lane * 2;
        b.q[0] = bls[0];
        b.q[1] = bls[1];
        acc[j]     = __builtin_amdgcn_wmma_f32_16x16x32_bf16(false, a0[cur].v, false, b.v,
                                                             (short)0, acc[j], false, false);
        acc[4 + j] = __builtin_amdgcn_wmma_f32_16x16x32_bf16(false, a1[cur].v, false, b.v,
                                                             (short)0, acc[4 + j], false, false);
      }
    }
  }
#pragma unroll
  for (int j = 0; j < 4; ++j) {
    int oc = (ng * 4 + j) * 16 + mrow;
    float bb = bmask[oc];
#pragma unroll
    for (int i = 0; i < 2; ++i) {
#pragma unroll
      for (int v = 0; v < 8; ++v) {
        int r = v + 8 * half;
        size_t pix = (size_t)n * HW + (size_t)y * W + x0 + i * 16 + r;
        float mval = 1.f / (1.f + __expf(-(acc[i * 4 + j][v] + bb)));
        float av = (float)align[pix * C5 + oc];
        afinal[pix * C5 + oc] = (bf16)(av * mval);
      }
    }
  }
}

// ---------------- final 1x1 conv GEMM (1Mx4N) + bias + relu -> NCHW f32 ----------------
__global__ void __launch_bounds__(32) k_gemm3(const bf16* __restrict__ A,
                                              const bf16* __restrict__ Bp,
                                              const float* __restrict__ bias,
                                              float* __restrict__ out) {
  int lane = threadIdx.x & 31;
  int ng = blockIdx.y;                  // 0..3 -> nt = ng*4 + j
  int mt = blockIdx.x % MT;
  int n  = blockIdx.x / MT;
  int half = lane >> 4, mrow = lane & 15;
  const u32x4* arow = (const u32x4*)(A + ((size_t)n * HW + mt * 16 + mrow) * C5);
  v8f acc[4] = {v8f{}, v8f{}, v8f{}, v8f{}};
  Frag a[2], b[2][4];
  auto load_step = [&](int ks, int buf) {
    const u32x4* ap = arow + ks * 4;
    a[buf].q[0] = ap[2 * half];
    a[buf].q[1] = ap[2 * half + 1];
#pragma unroll
    for (int j = 0; j < 4; ++j) {
      const u32x4* bp2 = (const u32x4*)Bp + ((size_t)((ng * 4 + j) * KS3 + ks) * 32 + lane) * 2;
      b[buf][j].q[0] = bp2[0];
      b[buf][j].q[1] = bp2[1];
    }
  };
  load_step(0, 0);
#pragma unroll
  for (int ks = 0; ks < KS3; ++ks) {
    int cur = ks & 1;
    if (ks + 1 < KS3) load_step(ks + 1, cur ^ 1);
#pragma unroll
    for (int j = 0; j < 4; ++j)
      acc[j] = __builtin_amdgcn_wmma_f32_16x16x32_bf16(false, a[cur].v, false, b[cur][j].v,
                                                       (short)0, acc[j], false, false);
  }
#pragma unroll
  for (int j = 0; j < 4; ++j) {
    int oc = (ng * 4 + j) * 16 + mrow;
    float bb = bias[oc];
#pragma unroll
    for (int v = 0; v < 8; ++v) {
      int r = v + 8 * half;
      out[((size_t)n * 256 + oc) * HW + (size_t)mt * 16 + r] = fmaxf(acc[j][v] + bb, 0.f);
    }
  }
}

extern "C" void kernel_launch(void* const* d_in, const int* in_sizes, int n_in,
                              void* d_out, int out_size, void* d_ws, size_t ws_size,
                              hipStream_t stream) {
  (void)in_sizes; (void)n_in; (void)out_size; (void)ws_size;
  const float* feature  = (const float*)d_in[0];
  const float* boxes    = (const float*)d_in[1];
  // d_in[2] = wh (unused by forward)
  const float* w_cur    = (const float*)d_in[3];
  // d_in[4] = b_cur  : cancels inside instance_norm
  const float* w_ltrb   = (const float*)d_in[5];
  // d_in[6] = b_ltrb : cancels inside instance_norm
  const float* w_mask   = (const float*)d_in[7];
  const float* b_mask   = (const float*)d_in[8];
  const float* w_border = (const float*)d_in[9];
  const float* b_border = (const float*)d_in[10];
  float* out = (float*)d_out;

  char* ws = (char*)d_ws;
  bf16*  featA   = (bf16*)(ws + OFF_FEATA);
  bf16*  B1p     = (bf16*)(ws + OFF_B1);
  bf16*  Bmaskp  = (bf16*)(ws + OFF_BMASK);
  bf16*  Bbordp  = (bf16*)(ws + OFF_BBORDER);
  float* C1      = (float*)(ws + OFF_C1);
  bf16*  Afinal  = (bf16*)(ws + OFF_C1);     // aliases C1 (dead after k_norm)
  float* stats   = (float*)(ws + OFF_STATS);
  bf16*  feat4   = (bf16*)(ws + OFF_FEAT4);
  bf16*  align   = (bf16*)(ws + OFF_ALIGN);

  // 1) feature NCHW f32 -> [N][HW][CIN] bf16
  k_feat_to_bf16<<<dim3(HW / 32, CIN / 32, NB), dim3(32, 8), 0, stream>>>(feature, featA);

  // 2) pack weight fragments
  {
    int t1 = NT1 * KS1 * 32 * 8;
    k_pack_b1<<<(t1 + 255) / 256, 256, 0, stream>>>(w_cur, w_ltrb, B1p);
    int t2 = NT1 * KS2 * 32 * 8;
    k_pack_bmask<<<(t2 + 255) / 256, 256, 0, stream>>>(w_mask, Bmaskp);
    int t3 = NT3 * KS3 * 32 * 8;
    k_pack_bborder<<<(t3 + 255) / 256, 256, 0, stream>>>(w_border, Bbordp);
  }

  // 3) fused cur+ltrb 1x1 conv (WMMA bf16, 1Mx4N per wave, double-buffered)
  k_gemm1<<<dim3(NB * MT, NT1 / 4), 32, 0, stream>>>(featA, B1p, C1);

  // 4) instance norm stats + normalize/relu
  k_stats<<<NB * C5, 256, 0, stream>>>(C1, stats);
  {
    size_t tot = (size_t)NB * HW * C5;
    k_norm<<<(unsigned)((tot + 255) / 256), 256, 0, stream>>>(C1, stats, align, feat4);
  }

  // 5) border align (scatter into align channels 0..511)
  k_border<<<NB * HW * 4, BC, 0, stream>>>(feat4, boxes, align);

  // 6) conv3x3 mask (WMMA + TDM-staged B in LDS) + sigmoid + multiply align
  k_conv3x3<<<dim3(NB * ((MT / 2) / 4), NT1 / 4), 128, 0, stream>>>(feat4, Bmaskp, b_mask,
                                                                    align, Afinal);

  // 7) final 1x1 conv (WMMA, 1Mx4N per wave, double-buffered) + bias + relu -> NCHW f32
  k_gemm3<<<dim3(NB * MT, NT3 / 4), 32, 0, stream>>>(Afinal, Bbordp, b_border, out);
}
